// CTRNN_41214506172739
// MI455X (gfx1250) — matrix operand order repaired and checked by reading, not compiled, not run on hardware
//
#include <hip/hip_runtime.h>
#include <hip/hip_bf16.h>

// CTRNN fused persistent-scan kernel for gfx1250 (wave32, WMMA).
//   T=1024, B=256, IN=128, H=512, O=32, alpha = dt/tau = 0.001
// 16 workgroups x 256 threads (8 wave32). WG g owns batch rows [16g, 16g+16).
// h-state lives in f32 WMMA accumulators (registers) + bf16 mirror in LDS.
// Residency policy (explicit, to avoid register-allocator scratch spills):
//   - W_inT fragments (16 x 8 VGPRs)      : register-resident across all t (LICM)
//   - W_rec / W_outT fragments            : streamed from L2 every step through a
//     software pipeline; their base pointers are laundered through an empty
//     asm() inside the t-loop so LICM CANNOT hoist them (no spills).

#define T_STEPS 1024
#define BATCH   256
#define IN_SZ   128
#define HID     512
#define OUT_SZ  32
#define ALPHA   0.001f
#define BT      16             // batch rows per workgroup
#define H_STR   (HID + 8)      // padded LDS stride (bf16 elems) -> conflict-free b128
#define X_STR   (IN_SZ + 8)
#define KC_REC  (HID / 32)     // 16 K-chunks in recurrent GEMM
#define KC_IN   (IN_SZ / 32)   // 4 K-chunks in input projection

typedef __attribute__((ext_vector_type(16))) __bf16 bf16x16;
typedef __attribute__((ext_vector_type(8)))  __bf16 bf16x8;
typedef __attribute__((ext_vector_type(8)))  float  f32x8;

// A-fragment (16x32 bf16, M x K) from LDS, contiguous k order (same as bfrag_g):
//   lanes 0-15 : row m=lane,    K = [k0, k0+16)
//   lanes 16-31: row m=lane-16, K = [k0+16, k0+32)
__device__ __forceinline__ bf16x16 afrag_lds(const __bf16* base, int stride, int k0, int lane) {
  const int m    = lane & 15;
  const int half = lane >> 4;           // 0 or 1
  const __bf16* p = base + m * stride + k0 + half * 16;
  bf16x8 c0 = *(const bf16x8*)(p);      // 16B ds_load_b128
  bf16x8 c1 = *(const bf16x8*)(p + 8);  // 16B ds_load_b128
  bf16x16 a;
#pragma unroll
  for (int i = 0; i < 8; ++i) { a[i] = c0[i]; a[i + 8] = c1[i]; }
  return a;
}

// B-fragment (32x16 bf16, K x N). B column n is stored as contiguous row n of W
// (W_rec rows == columns of W_rec^T; W_in/W_out are pre-transposed in prep).
//   lanes 0-15 : col n=lane,    K = [k0, k0+16)
//   lanes 16-31: col n=lane-16, K = [k0+16, k0+32)
__device__ __forceinline__ bf16x16 bfrag_g(const __bf16* __restrict__ W, int rowlen,
                                           int n0, int k0, int lane) {
  const __bf16* p = W + (size_t)(n0 + (lane & 15)) * rowlen + k0 + (lane >> 4) * 16;
  return *(const bf16x16*)p;            // 32B contiguous (2x global_load_b128)
}

// ---------------- prep: f32 weights -> bf16 (W_in/W_out transposed) -------------
__global__ void ctrnn_prep(const float* __restrict__ Wrec, const float* __restrict__ Win,
                           const float* __restrict__ Wout,
                           __bf16* __restrict__ wrec_b, __bf16* __restrict__ winT_b,
                           __bf16* __restrict__ woutT_b) {
  int i = blockIdx.x * blockDim.x + threadIdx.x;
  if (i < HID * HID) { wrec_b[i] = (__bf16)Wrec[i]; return; }
  i -= HID * HID;
  if (i < HID * IN_SZ) {                 // W_inT[n][k] = W_in[k][n]
    int n = i / IN_SZ, k = i % IN_SZ;
    winT_b[i] = (__bf16)Win[k * HID + n];
    return;
  }
  i -= HID * IN_SZ;
  if (i < OUT_SZ * HID) {                // W_outT[o][n] = W_out[n][o]
    int o = i / HID, n = i % HID;
    woutT_b[i] = (__bf16)Wout[n * OUT_SZ + o];
  }
}

// ---------------- fused persistent scan ----------------------------------------
__global__ __launch_bounds__(256, 1)
void ctrnn_scan(const float* __restrict__ inputs, const float* __restrict__ b_rec,
                const float* __restrict__ b_out,
                const __bf16* __restrict__ wrec_b, const __bf16* __restrict__ winT_b,
                const __bf16* __restrict__ woutT_b, float* __restrict__ out) {
  __shared__ __bf16 h_lds[BT * H_STR];   // bf16 mirror of this WG's h tile
  __shared__ __bf16 x_lds[BT * X_STR];   // current input tile, bf16

  const int tid   = threadIdx.x;
  const int lane  = tid & 31;
  const int wave  = tid >> 5;            // 8 waves
  const int b0    = blockIdx.x * BT;     // batch-row base
  const int n0    = wave * 64;           // this wave's 64 hidden columns
  const int lcol  = lane & 15;
  const int lhalf = lane >> 4;

  for (int i = tid; i < BT * H_STR; i += 256) h_lds[i] = (__bf16)0.0f;

  f32x8 hstate[4];                       // f32 h-state, exactly the WMMA C layout
  float brec[4];
#pragma unroll
  for (int j = 0; j < 4; ++j) {
    brec[j] = b_rec[n0 + j * 16 + lcol];
#pragma unroll
    for (int e = 0; e < 8; ++e) hstate[j][e] = 0.0f;
  }
  const int o0 = (wave - 6) * 16;        // output-proj tile for waves 6,7
  const float bo = (wave >= 6) ? b_out[o0 + lcol] : 0.0f;

  const int srow = tid >> 4;             // staging: row 0..15
  const int scol = (tid & 15) * 8;       // 8 floats per thread
  __syncthreads();

  for (int t = 0; t < T_STEPS; ++t) {
    // Launder streamed-weight pointers: the empty asm makes them loop-variant,
    // so LICM cannot hoist their (t-invariant) loads -> no scratch spills.
    const __bf16* wrec_s  = wrec_b;
    const __bf16* woutT_s = woutT_b;
    asm volatile("" : "+s"(wrec_s), "+s"(woutT_s));

    // (a) stage inputs[t, btile, :] -> bf16 LDS (coalesced 32B f32 loads)
    {
      const float* ip = inputs + ((size_t)t * BATCH + (b0 + srow)) * IN_SZ + scol;
      __bf16* xp = x_lds + srow * X_STR + scol;
#pragma unroll
      for (int q = 0; q < 8; ++q) xp[q] = (__bf16)ip[q];
    }
    __syncthreads();  // B1: x ready; previous step fully consumed

    // (b) acc = b_rec + x @ W_in + h @ W_rec^T   (cols [n0, n0+64))
    f32x8 acc[4];
#pragma unroll
    for (int j = 0; j < 4; ++j)
#pragma unroll
      for (int e = 0; e < 8; ++e) acc[j][e] = brec[j];

    // ---- input projection: preload all 4 A-frags (one batched dscnt wait),
    //      W_in B-frags stay register-resident across the whole scan (LICM) ----
    {
      bf16x16 xa[KC_IN];
#pragma unroll
      for (int kc = 0; kc < KC_IN; ++kc)
        xa[kc] = afrag_lds(x_lds, X_STR, kc * 32, lane);
#pragma unroll
      for (int kc = 0; kc < KC_IN; ++kc)
#pragma unroll
        for (int j = 0; j < 4; ++j) {
          bf16x16 b = bfrag_g(winT_b, IN_SZ, n0 + j * 16, kc * 32, lane);
          acc[j] = __builtin_amdgcn_wmma_f32_16x16x32_bf16(false, xa[kc], false, b,
                                                           (short)0, acc[j], false, false);
        }
    }

    // ---- recurrent GEMM: B stream depth-3 (4-buffer rotation), A depth-2
    //      (3-buffer). Chunks kc..kc+2 of W_rec are in flight before kc is
    //      consumed -> partial s_wait_loadcnt instead of full drains. ----
    {
      bf16x16 aa[3];
      bf16x16 bb[4][4];
#pragma unroll
      for (int p = 0; p < 3; ++p)            // B prologue: chunks 0,1,2
#pragma unroll
        for (int j = 0; j < 4; ++j)
          bb[p][j] = bfrag_g(wrec_s, HID, n0 + j * 16, p * 32, lane);
#pragma unroll
      for (int p = 0; p < 2; ++p)            // A prologue: chunks 0,1
        aa[p] = afrag_lds(h_lds, H_STR, p * 32, lane);

#pragma unroll
      for (int kc = 0; kc < KC_REC; ++kc) {
        const int bcur = kc & 3;
        const int bnxt = (kc + 3) & 3;       // freed at kc-1
        const int acur = kc % 3;
        const int anxt = (kc + 2) % 3;
        if (kc + 3 < KC_REC)                 // refill B for kc+3
#pragma unroll
          for (int j = 0; j < 4; ++j)
            bb[bnxt][j] = bfrag_g(wrec_s, HID, n0 + j * 16, (kc + 3) * 32, lane);
        if (kc + 2 < KC_REC)                 // refill A for kc+2
          aa[anxt] = afrag_lds(h_lds, H_STR, (kc + 2) * 32, lane);
#pragma unroll
        for (int j = 0; j < 4; ++j)
          acc[j] = __builtin_amdgcn_wmma_f32_16x16x32_bf16(false, aa[acur], false, bb[bcur][j],
                                                           (short)0, acc[j], false, false);
      }
    }
    __syncthreads();  // B2: all reads of old h done

    // (c) h = (1-a)*h + a*relu(acc); f32 in regs, bf16 mirror to LDS
#pragma unroll
    for (int j = 0; j < 4; ++j) {
      const int col = n0 + j * 16 + lcol;
#pragma unroll
      for (int e = 0; e < 8; ++e) {
        float pre = acc[j][e];
        float hn  = (1.0f - ALPHA) * hstate[j][e] + ALPHA * (pre > 0.0f ? pre : 0.0f);
        hstate[j][e] = hn;
        h_lds[(lhalf * 8 + e) * H_STR + col] = (__bf16)hn;
      }
    }
    __syncthreads();  // B3: new h visible

    // (d) waves 6,7: out[t] = h_new @ W_out + b_out (other waves roll into t+1).
    //     Depth-2 pipelines on both streams; W_outT pointer is laundered too.
    if (wave >= 6) {
      f32x8 oacc;
#pragma unroll
      for (int e = 0; e < 8; ++e) oacc[e] = bo;
      bf16x16 oaa[3], obb[3];
#pragma unroll
      for (int p = 0; p < 2; ++p) {
        oaa[p] = afrag_lds(h_lds, H_STR, p * 32, lane);
        obb[p] = bfrag_g(woutT_s, HID, o0, p * 32, lane);
      }
#pragma unroll
      for (int kc = 0; kc < KC_REC; ++kc) {
        const int cur = kc % 3;
        const int nxt = (kc + 2) % 3;
        if (kc + 2 < KC_REC) {
          oaa[nxt] = afrag_lds(h_lds, H_STR, (kc + 2) * 32, lane);
          obb[nxt] = bfrag_g(woutT_s, HID, o0, (kc + 2) * 32, lane);
        }
        oacc = __builtin_amdgcn_wmma_f32_16x16x32_bf16(false, oaa[cur], false, obb[cur],
                                                       (short)0, oacc, false, false);
      }
      float* op = out + (size_t)t * (BATCH * OUT_SZ);
#pragma unroll
      for (int e = 0; e < 8; ++e) {
        int m = lhalf * 8 + e;
        op[(b0 + m) * OUT_SZ + o0 + lcol] = oacc[e];
      }
    }
  }
}

extern "C" void kernel_launch(void* const* d_in, const int* in_sizes, int n_in,
                              void* d_out, int out_size, void* d_ws, size_t ws_size,
                              hipStream_t stream) {
  const float* inputs = (const float*)d_in[0];   // [1024][256][128]
  const float* W_rec  = (const float*)d_in[1];   // [512][512]
  const float* W_in   = (const float*)d_in[2];   // [128][512]
  const float* b_rec  = (const float*)d_in[3];   // [512]
  const float* W_out  = (const float*)d_in[4];   // [512][32]
  const float* b_out  = (const float*)d_in[5];   // [32]
  float* out = (float*)d_out;                    // [1024][256][32]

  __bf16* wrec_b  = (__bf16*)d_ws;               // 512*512
  __bf16* winT_b  = wrec_b + HID * HID;          // 512*128 (transposed)
  __bf16* woutT_b = winT_b + HID * IN_SZ;        // 32*512  (transposed)

  const int prep_elems = HID * HID + HID * IN_SZ + OUT_SZ * HID;
  ctrnn_prep<<<(prep_elems + 255) / 256, 256, 0, stream>>>(W_rec, W_in, W_out,
                                                           wrec_b, winT_b, woutT_b);
  ctrnn_scan<<<BATCH / BT, 256, 0, stream>>>(inputs, b_rec, b_out,
                                             wrec_b, winT_b, woutT_b, out);
}
